// cont_shGLM_50448685859422
// MI455X (gfx1250) — matrix operand order repaired and checked by reading, not compiled, not run on hardware
//
#include <hip/hip_runtime.h>

typedef __attribute__((ext_vector_type(16))) _Float16 v16h;
typedef __attribute__((ext_vector_type(8)))  float    v8f;

#define T_DATA 50000
#define E_NO   800
#define I_NO   200
#define SUBN   8
#define TSYN   200
#define THIST  100

__device__ __forceinline__ _Float16 h16(float x) { return (_Float16)x; }

// wave-uniform lane broadcast (source lane is a compile-time constant)
__device__ __forceinline__ float lanef(float v, int l) {
  return __int_as_float(__builtin_amdgcn_readlane(__float_as_int(v), l));
}

// ---------------------------------------------------------------------------
// Kernel 0: build temporal synaptic kernels kern_e / kern_i  (8 x 200 each)
// ---------------------------------------------------------------------------
__global__ __launch_bounds__(256) void k_build(const float* __restrict__ K_syn,
                                               const float* __restrict__ tau_syn,
                                               const float* __restrict__ delta_syn,
                                               float* __restrict__ kern_e,
                                               float* __restrict__ kern_i) {
  int id = blockIdx.x * 256 + threadIdx.x;
  if (id >= SUBN * TSYN) return;
  int s = id / TSYN, t = id % TSYN;
  float de = __expf(delta_syn[s * 2 + 0]);
  float di = __expf(delta_syn[s * 2 + 1]);
  float te = fmaxf((float)t - de, 0.f);
  float ti = fmaxf((float)t - di, 0.f);
  float acc_e = 0.f, acc_i = 0.f;
#pragma unroll
  for (int b = 0; b < 2; ++b) {
    float taue = __expf(tau_syn[b * 3 + 0]);
    float taui = __expf(tau_syn[b * 3 + 1]);
    float xe = te / taue;
    float xi = ti / taui;
    acc_e += K_syn[(s * 2 + b) * 3 + 0] * xe * __expf(-xe);
    acc_i += K_syn[(s * 2 + b) * 3 + 1] * xi * __expf(-xi);
  }
  kern_e[id] = acc_e;
  kern_i[id] = acc_i;
}

// ---------------------------------------------------------------------------
// Kernel 1: spike aggregation as WMMA GEMM:  out[t,n] = sum_k S[t,k]*C[n,k]
// K is a compile-time constant -> main loop is guard-free with b128 loads.
// A (16x32 f16): lanes 0-15 M=lane, elems 0..15 -> K kb+0..15
//                lanes 16-31 M=lane-16,       K kb+16..31.
// B (32x16 f16): N = lane&15, elem i -> K = kb + 16*(lane>>4) + i.
// D (16x16 f32): elem v -> row M = v + 8*(lane>>4), col N = lane&15.
// ---------------------------------------------------------------------------
template <int K>
__global__ __launch_bounds__(32) void k_agg_wmma(const float* __restrict__ S,
                                                 const float* __restrict__ C,
                                                 float* __restrict__ outv) {
  const int lane = threadIdx.x;
  const int hi   = lane >> 4;
  const int l16  = lane & 15;
  const int row  = blockIdx.x * 16 + l16;
  const bool bvalid = (l16 < SUBN);
  const float* __restrict__ rowp = S + (size_t)row * K;
  const float* __restrict__ colp = C + (size_t)(bvalid ? l16 : 0) * K;

  v8f acc = {};
  constexpr int KFULL = (K / 32) * 32;

  for (int kb = 0; kb < KFULL; kb += 32) {
    const float4* pa0 = (const float4*)(rowp + kb + hi * 16);
    const float4* pa1 = (const float4*)(rowp + kb + hi * 16 + 8);
    const float4* pb  = (const float4*)(colp + kb + hi * 16);
    float4 a0 = pa0[0], a1 = pa0[1];
    float4 a2 = pa1[0], a3 = pa1[1];
    float4 b0 = pb[0],  b1 = pb[1], b2 = pb[2], b3 = pb[3];

    v16h a, b;
    a[0]  = h16(a0.x); a[1]  = h16(a0.y); a[2]  = h16(a0.z); a[3]  = h16(a0.w);
    a[4]  = h16(a1.x); a[5]  = h16(a1.y); a[6]  = h16(a1.z); a[7]  = h16(a1.w);
    a[8]  = h16(a2.x); a[9]  = h16(a2.y); a[10] = h16(a2.z); a[11] = h16(a2.w);
    a[12] = h16(a3.x); a[13] = h16(a3.y); a[14] = h16(a3.z); a[15] = h16(a3.w);

    const _Float16 z0 = (_Float16)0.f;
    b[0]  = bvalid ? h16(b0.x) : z0; b[1]  = bvalid ? h16(b0.y) : z0;
    b[2]  = bvalid ? h16(b0.z) : z0; b[3]  = bvalid ? h16(b0.w) : z0;
    b[4]  = bvalid ? h16(b1.x) : z0; b[5]  = bvalid ? h16(b1.y) : z0;
    b[6]  = bvalid ? h16(b1.z) : z0; b[7]  = bvalid ? h16(b1.w) : z0;
    b[8]  = bvalid ? h16(b2.x) : z0; b[9]  = bvalid ? h16(b2.y) : z0;
    b[10] = bvalid ? h16(b2.z) : z0; b[11] = bvalid ? h16(b2.w) : z0;
    b[12] = bvalid ? h16(b3.x) : z0; b[13] = bvalid ? h16(b3.y) : z0;
    b[14] = bvalid ? h16(b3.z) : z0; b[15] = bvalid ? h16(b3.w) : z0;

    acc = __builtin_amdgcn_wmma_f32_16x16x32_f16(
        false, a, false, b, (short)0, acc, false, false);
  }

  constexpr int REM = K - KFULL;
  if (REM > 0) {
    // compile-time element guards; hi-dependent placement via selects
    float av[32], bv[32];
#pragma unroll
    for (int i = 0; i < 32; ++i) {
      av[i] = (i < REM) ? rowp[KFULL + i] : 0.f;
      bv[i] = (i < REM) ? colp[KFULL + i] : 0.f;
    }
    v16h a, b;
    const _Float16 z0 = (_Float16)0.f;
#pragma unroll
    for (int i = 0; i < 8; ++i) {
      a[i]     = hi ? h16(av[16 + i]) : h16(av[i]);       // K = KFULL + hi*16 + i
      a[i + 8] = hi ? h16(av[24 + i]) : h16(av[8 + i]);   // K = KFULL + hi*16 + 8 + i
    }
#pragma unroll
    for (int i = 0; i < 16; ++i) {
      _Float16 val = hi ? h16(bv[16 + i]) : h16(bv[i]);   // K = KFULL + hi*16 + i
      b[i] = bvalid ? val : z0;
    }
    acc = __builtin_amdgcn_wmma_f32_16x16x32_f16(
        false, a, false, b, (short)0, acc, false, false);
  }

  if (bvalid) {
#pragma unroll
    for (int v = 0; v < 8; ++v) {
      int r = blockIdx.x * 16 + v + hi * 8;
      outv[(size_t)r * SUBN + l16] = acc[v];
    }
  }
}

// ---------------------------------------------------------------------------
// Kernel 2: causal depthwise convolution, 200 taps, 8 channels.
// ---------------------------------------------------------------------------
__global__ __launch_bounds__(256) void k_conv(const float* __restrict__ in_e,
                                              const float* __restrict__ in_i,
                                              const float* __restrict__ ke,
                                              const float* __restrict__ ki,
                                              float* __restrict__ syn) {
  __shared__ float ske[SUBN * TSYN];
  __shared__ float ski[SUBN * TSYN];
  for (int i = threadIdx.x; i < SUBN * TSYN; i += 256) {
    ske[i] = ke[i];
    ski[i] = ki[i];
  }
  __syncthreads();
  int id = blockIdx.x * 256 + threadIdx.x;
  if (id >= T_DATA * SUBN) return;
  int t = id >> 3, s = id & 7;
  const float* kes = ske + s * TSYN;
  const float* kis = ski + s * TSYN;
  float acc = 0.f;
  int dmax = (t + 1 < TSYN) ? (t + 1) : TSYN;
  for (int d = 0; d < dmax; ++d) {
    size_t idx = (size_t)(t - d) * SUBN + s;
    acc += in_e[idx] * kes[d] + in_i[idx] * kis[d];
  }
  syn[id] = acc;
}

// ---------------------------------------------------------------------------
// Kernel 3: sequential scan, single wave32. Lane s = subunit s.
// History convolutions via exact finite-window linear recurrences:
//   Q' = z_t + r*Q - z_{t-100}*r^100
//   P' = r*(P+Q) - z_{t-100}*100*r^100
//   fh[k] = sum_b K_spike[k,b]/tauH_b * PH_b[k]
//   fa[s] = sum_b K_syn[s,b,2]/tauA_b * sum_c C_den[s,c]*PA_b[c-1]
// Cross-lane couplings use v_readlane (uniform source lane).
// ---------------------------------------------------------------------------
__global__ __launch_bounds__(32) void k_scan(
    const float* __restrict__ syn, const float* __restrict__ noise,
    const float* __restrict__ C_den, const float* __restrict__ V_o,
    const float* __restrict__ K_syn, const float* __restrict__ tau_syn,
    const float* __restrict__ W_sub, const float* __restrict__ theta_syn,
    const float* __restrict__ theta_spike, const float* __restrict__ W_spike,
    const float* __restrict__ tau_hist, const float* __restrict__ K_spike,
    float* __restrict__ out) {
  __shared__ float ring[THIST * SUBN];
  const int lane = threadIdx.x;
  const int s = lane & 7;
  const bool zlane = (lane >= 1 && lane <= 7);
  const int k = zlane ? (lane - 1) : 0;

  for (int i = lane; i < THIST * SUBN; i += 32) ring[i] = 0.f;
  __syncthreads();

  const float th_syn = theta_syn[s];
  const float wsub = W_sub[s];
  const float vo = V_o[0];
  float cd[8];
#pragma unroll
  for (int c = 0; c < 8; ++c) cd[c] = C_den[s * 8 + c];

  const float tauH0 = __expf(tau_hist[0]);
  const float tauH1 = __expf(tau_hist[1]);
  const float rH0 = __expf(-1.f / tauH0), rH1 = __expf(-1.f / tauH1);
  const float rH0w = __expf(-100.f / tauH0), rH1w = __expf(-100.f / tauH1);
  const float cH0 = zlane ? K_spike[k * 2 + 0] / tauH0 : 0.f;
  const float cH1 = zlane ? K_spike[k * 2 + 1] / tauH1 : 0.f;

  const float tauA0 = __expf(tau_syn[0 * 3 + 2]);
  const float tauA1 = __expf(tau_syn[1 * 3 + 2]);
  const float rA0 = __expf(-1.f / tauA0), rA1 = __expf(-1.f / tauA1);
  const float rA0w = __expf(-100.f / tauA0), rA1w = __expf(-100.f / tauA1);
  const float cA0 = K_syn[(s * 2 + 0) * 3 + 2] / tauA0;
  const float cA1 = K_syn[(s * 2 + 1) * 3 + 2] / tauA1;

  const float wspk = zlane ? W_spike[k] : 0.f;
  const float thspk = zlane ? theta_spike[k] : 0.f;

  float y_tm1 = 0.f, y_t = 0.f;
  float QH0 = 0, PH0 = 0, QH1 = 0, PH1 = 0;
  float QA0 = 0, PA0 = 0, QA1 = 0, PA1 = 0;

  float syn_cur = syn[s];
  float nse_cur = zlane ? noise[k] : 0.f;

  float* __restrict__ outY = out + T_DATA;
  float* __restrict__ outZ = out + T_DATA + (size_t)T_DATA * 7;

  for (int t = 0; t < T_DATA; ++t) {
    float syn_nxt = 0.f, nse_nxt = 0.f;
    if (t + 1 < T_DATA) {
      syn_nxt = syn[(size_t)(t + 1) * SUBN + s];
      if (zlane) nse_nxt = noise[(size_t)(t + 1) * 7 + k];
    }

    float fh = cH0 * PH0 + cH1 * PH1;  // == 0 on lane 0 by construction

    float dend = 0.f, sA0 = 0.f, sA1 = 0.f;
#pragma unroll
    for (int c = 0; c < 8; ++c) {
      float w = cd[c];
      dend += w * lanef(y_tm1, c);
      if (c >= 1) {
        sA0 += w * lanef(PA0, c);
        sA1 += w * lanef(PA1, c);
      }
    }
    float fa = cA0 * sA0 + cA1 * sA1;

    float x_in = syn_cur + fa + dend + th_syn + fh;
    float x_out = 1.f / (1.f + __expf(-x_in));
    float y_tp1 = x_out * wsub;
    float z = 1.f / (1.f + __expf(-(x_out * wspk + thspk + 100.f * nse_cur)));

    float z_old = 0.f;
    int rp = (t % THIST) * SUBN + s;
    if (zlane) {
      z_old = ring[rp];
      ring[rp] = z;
    }

    float nQ;
    nQ = z + rH0 * QH0 - z_old * rH0w;
    PH0 = rH0 * (PH0 + QH0) - z_old * (100.f * rH0w); QH0 = nQ;
    nQ = z + rH1 * QH1 - z_old * rH1w;
    PH1 = rH1 * (PH1 + QH1) - z_old * (100.f * rH1w); QH1 = nQ;
    nQ = z + rA0 * QA0 - z_old * rA0w;
    PA0 = rA0 * (PA0 + QA0) - z_old * (100.f * rA0w); QA0 = nQ;
    nQ = z + rA1 * QA1 - z_old * rA1w;
    PA1 = rA1 * (PA1 + QA1) - z_old * (100.f * rA1w); QA1 = nQ;

    if (lane == 0) out[t] = y_tp1 + vo;
    if (zlane) {
      outY[(size_t)t * 7 + k] = y_tp1;
      outZ[(size_t)t * 7 + k] = z;
    }

    y_tm1 = y_t;
    y_t = y_tp1;
    syn_cur = syn_nxt;
    nse_cur = nse_nxt;
  }
}

// ---------------------------------------------------------------------------
extern "C" void kernel_launch(void* const* d_in, const int* in_sizes, int n_in,
                              void* d_out, int out_size, void* d_ws,
                              size_t ws_size, hipStream_t stream) {
  (void)in_sizes; (void)n_in; (void)out_size; (void)ws_size;
  const float* S_e       = (const float*)d_in[0];
  const float* S_i       = (const float*)d_in[1];
  const float* noise     = (const float*)d_in[2];
  const float* C_den     = (const float*)d_in[3];
  const float* C_syn_e   = (const float*)d_in[4];
  const float* C_syn_i   = (const float*)d_in[5];
  const float* V_o       = (const float*)d_in[6];
  const float* K_syn     = (const float*)d_in[7];
  const float* tau_syn   = (const float*)d_in[8];
  const float* delta_syn = (const float*)d_in[9];
  const float* W_sub     = (const float*)d_in[10];
  const float* theta_syn = (const float*)d_in[11];
  const float* theta_spk = (const float*)d_in[12];
  const float* W_spike   = (const float*)d_in[13];
  const float* tau_hist  = (const float*)d_in[14];
  const float* K_spike   = (const float*)d_in[15];
  float* out = (float*)d_out;

  float* w      = (float*)d_ws;
  float* in_e   = w;            // 400000 floats
  float* in_i   = w + 400000;   // 400000
  float* syn    = w + 800000;   // 400000
  float* kern_e = w + 1200000;  // 1600
  float* kern_i = w + 1201600;  // 1600

  k_build<<<(SUBN * TSYN + 255) / 256, 256, 0, stream>>>(K_syn, tau_syn,
                                                         delta_syn, kern_e,
                                                         kern_i);
  k_agg_wmma<E_NO><<<T_DATA / 16, 32, 0, stream>>>(S_e, C_syn_e, in_e);
  k_agg_wmma<I_NO><<<T_DATA / 16, 32, 0, stream>>>(S_i, C_syn_i, in_i);
  k_conv<<<(T_DATA * SUBN + 255) / 256, 256, 0, stream>>>(in_e, in_i, kern_e,
                                                          kern_i, syn);
  k_scan<<<1, 32, 0, stream>>>(syn, noise, C_den, V_o, K_syn, tau_syn, W_sub,
                               theta_syn, theta_spk, W_spike, tau_hist,
                               K_spike, out);
}
